// MistralEmbeddingModel_11742440587782
// MI455X (gfx1250) — compile-verified, matrix operand order checked by reading, not run on hardware
//
#include <hip/hip_runtime.h>

// ---------------------------------------------------------------------------
// CDNA5 (gfx1250) QLoRA transformer forward.  bf16 WMMA (16x16x32) everywhere.
// v3: - K templated into the GEMM kernels (compile-time trip counts ->
//       unrolling / software pipelining across K-steps)
//     - A-fragments hoisted into named values so all global_load_b128s issue
//       before the 4-WMMA cluster (load latency hiding, staggered waits)
//     - NF4 lookup table staged in LDS (ds_load, conflict-free) instead of
//       divergent per-lane constant-buffer gathers through VMEM
// ---------------------------------------------------------------------------

typedef __bf16 bf16;
typedef __attribute__((ext_vector_type(16))) bf16  v16bf;
typedef __attribute__((ext_vector_type(8)))  float v8f;

constexpr int L_  = 2;
constexpr int D_  = 2048;
constexpr int H_  = 16;
constexpr int HK_ = 4;
constexpr int HD_ = 128;
constexpr int F_  = 8192;
constexpr int R_  = 64;
constexpr int B_  = 16;
constexpr int S_  = 128;
constexpr int N_  = B_ * S_;        // 2048 tokens
constexpr int KVD_ = HK_ * HD_;     // 512
constexpr float SCALE_ = 2.0f;      // 128/64
constexpr float EPS_   = 1e-5f;

__constant__ float NF4_TAB[16] = {
    -1.0f, -0.6961928009986877f, -0.5250730514526367f, -0.39491748809814453f,
    -0.28444138169288635f, -0.18477343022823334f, -0.09105003625154495f, 0.0f,
    0.07958029955625534f, 0.16093020141124725f, 0.24611230194568634f,
    0.33791524171829224f, 0.44070982933044434f, 0.5626170039176941f,
    0.7229568362236023f, 1.0f};

struct BF8  { bf16  v[8]; };   // 16-byte chunk of 8 bf16
struct F8   { float v[8]; };   // 32-byte chunk of 8 floats
struct I8x  { int   v[8]; };   // 32-byte chunk of 8 int32 codes

__device__ inline v8f wmma_bf16(v16bf a, v16bf b, v8f c) {
  // v_wmma_f32_16x16x32_bf16  (neg_a, A, neg_b, B, c_mod, C, reuse_a, reuse_b)
  return __builtin_amdgcn_wmma_f32_16x16x32_bf16(false, a, false, b,
                                                 (short)0, c, false, false);
}

// A-fragment (or K-row B-fragment) from a row-major bf16 matrix.
// Lane l holds row (row0 + l&15); element e maps to
//   k = k0 + (e<8?0:16) + 8*(l>>4) + (e&7)      (ISA 7.12.2, 16-bit A 16x32)
__device__ inline v16bf frag_bf16(const bf16* __restrict__ base, int ld,
                                  int row0, int k0, int lane) {
  const long r  = row0 + (lane & 15);
  const int  kb = k0 + 8 * (lane >> 4);
  BF8 lo = *(const BF8*)(base + r * ld + kb);
  BF8 hi = *(const BF8*)(base + r * ld + kb + 16);
  v16bf f;
#pragma unroll
  for (int e = 0; e < 8; ++e) { f[e] = lo.v[e]; f[e + 8] = hi.v[e]; }
  return f;
}

// Same fragment, sourced from a row-major f32 matrix (LoRA A / B factors).
__device__ inline v16bf frag_f32(const float* __restrict__ base, int ld,
                                 int row0, int k0, int lane) {
  const long r  = row0 + (lane & 15);
  const int  kb = k0 + 8 * (lane >> 4);
  F8 lo = *(const F8*)(base + r * ld + kb);
  F8 hi = *(const F8*)(base + r * ld + kb + 16);
  v16bf f;
#pragma unroll
  for (int e = 0; e < 8; ++e) {
    f[e]     = (bf16)lo.v[e];
    f[e + 8] = (bf16)hi.v[e];
  }
  return f;
}

// B-fragment from NF4 codes: dequantize 16 codes/lane in registers.
// k0 is a multiple of 32 so the whole 32-wide K chunk shares one absmax block.
// The 16-entry NF4 table lives in LDS (ds_load, 16 addrs over 64 banks).
__device__ inline v16bf frag_nf4(const int* __restrict__ codes,
                                 const float* __restrict__ absmax,
                                 const float* tab,  // LDS
                                 int K, int o0, int k0, int lane) {
  const long o  = o0 + (lane & 15);
  const int  kb = k0 + 8 * (lane >> 4);
  const float am = absmax[o * (K >> 6) + (k0 >> 6)];
  I8x c0 = *(const I8x*)(codes + o * K + kb);
  I8x c1 = *(const I8x*)(codes + o * K + kb + 16);
  v16bf f;
#pragma unroll
  for (int e = 0; e < 8; ++e) {
    f[e]     = (bf16)(tab[c0.v[e] & 15] * am);
    f[e + 8] = (bf16)(tab[c1.v[e] & 15] * am);
  }
  return f;
}

// ---------------------------------------------------------------------------
// Kernels
// ---------------------------------------------------------------------------

__global__ __launch_bounds__(256) void k_embed(const int* __restrict__ ids,
                                               const float* __restrict__ emb,
                                               float* __restrict__ h) {
  const long tok = blockIdx.x;
  const long id  = ids[tok];
  const float4* src = (const float4*)(emb + id * D_);
  float4* dst = (float4*)(h + tok * D_);
  for (int i = threadIdx.x; i < D_ / 4; i += 256) dst[i] = src[i];
}

__global__ __launch_bounds__(256) void k_rmsnorm(const float* __restrict__ h,
                                                 const float* __restrict__ w,
                                                 bf16* __restrict__ xb) {
  const long row = blockIdx.x;
  const float* x = h + row * D_;
  __shared__ float red[256];
  float s = 0.f;
  for (int i = threadIdx.x; i < D_; i += 256) { float v = x[i]; s += v * v; }
  red[threadIdx.x] = s;
  __syncthreads();
  for (int off = 128; off > 0; off >>= 1) {
    if (threadIdx.x < off) red[threadIdx.x] += red[threadIdx.x + off];
    __syncthreads();
  }
  const float inv = rsqrtf(red[0] / (float)D_ + EPS_);
  for (int i = threadIdx.x; i < D_; i += 256)
    xb[row * D_ + i] = (bf16)(x[i] * inv * w[i]);
}

// XA = X(bf16 N x K) @ A(f32 64 x K)^T   -> bf16 [N,64]
// Block = 4 waves; wave w owns o-tile w*16; each wave covers 64 M rows so the
// converted A-factor fragment is reused by 4 WMMAs.
template <int K>
__global__ __launch_bounds__(128) void k_lora_xa(const bf16* __restrict__ X,
                                                 const float* __restrict__ A,
                                                 bf16* __restrict__ Y) {
  const int lane = threadIdx.x & 31, wave = threadIdx.x >> 5;
  const int o0 = wave * 16;            // 4 waves cover R_=64
  const int m0 = blockIdx.x * 64;
  v8f acc0 = {}, acc1 = {}, acc2 = {}, acc3 = {};
  for (int k0 = 0; k0 < K; k0 += 32) {
    v16bf b  = frag_f32(A, K, o0, k0, lane);
    v16bf a0 = frag_bf16(X, K, m0,      k0, lane);
    v16bf a1 = frag_bf16(X, K, m0 + 16, k0, lane);
    v16bf a2 = frag_bf16(X, K, m0 + 32, k0, lane);
    v16bf a3 = frag_bf16(X, K, m0 + 48, k0, lane);
    acc0 = wmma_bf16(a0, b, acc0);
    acc1 = wmma_bf16(a1, b, acc1);
    acc2 = wmma_bf16(a2, b, acc2);
    acc3 = wmma_bf16(a3, b, acc3);
  }
  const int n = o0 + (lane & 15);
  v8f accs[4] = {acc0, acc1, acc2, acc3};
#pragma unroll
  for (int mt = 0; mt < 4; ++mt) {
    const int mrow = m0 + mt * 16 + 8 * (lane >> 4);
#pragma unroll
    for (int r = 0; r < 8; ++r)
      Y[(long)(mrow + r) * R_ + n] = (bf16)accs[mt][r];
  }
}

// Y(f32 N x O) = X @ dequant_nf4(W[O,K])^T + SCALE * XA[N,64] @ Bm[O,64]^T
// Block = 4 waves; wave w owns o-tile (blockIdx.x*64 + w*16) and iterates
// 4 M tiles, reusing each dequantized weight fragment for 4 WMMAs.
template <int K>
__global__ __launch_bounds__(128) void k_qlinear(
    const bf16* __restrict__ X, const int* __restrict__ codes,
    const float* __restrict__ absmax, const bf16* __restrict__ XA,
    const float* __restrict__ Bm, float* __restrict__ Y, int O) {
  __shared__ float tab[16];
  if (threadIdx.x < 16) tab[threadIdx.x] = NF4_TAB[threadIdx.x];
  __syncthreads();

  const int lane = threadIdx.x & 31, wave = threadIdx.x >> 5;
  const int o0 = blockIdx.x * 64 + wave * 16;
  const int m0 = blockIdx.y * 64;
  v8f acc0 = {}, acc1 = {}, acc2 = {}, acc3 = {};
  for (int k0 = 0; k0 < K; k0 += 32) {
    // keep the NF4 code stream ahead of the WMMA pipe (global_prefetch_b8)
    if ((k0 & 255) == 0 && k0 + 256 < K) {
      const long o = o0 + (lane & 15);
      __builtin_prefetch(codes + o * (long)K + k0 + 256, 0, 1);
    }
    v16bf w  = frag_nf4(codes, absmax, tab, K, o0, k0, lane);
    v16bf a0 = frag_bf16(X, K, m0,      k0, lane);
    v16bf a1 = frag_bf16(X, K, m0 + 16, k0, lane);
    v16bf a2 = frag_bf16(X, K, m0 + 32, k0, lane);
    v16bf a3 = frag_bf16(X, K, m0 + 48, k0, lane);
    acc0 = wmma_bf16(a0, w, acc0);
    acc1 = wmma_bf16(a1, w, acc1);
    acc2 = wmma_bf16(a2, w, acc2);
    acc3 = wmma_bf16(a3, w, acc3);
  }
  v8f lacc0 = {}, lacc1 = {}, lacc2 = {}, lacc3 = {};
#pragma unroll
  for (int k0 = 0; k0 < R_; k0 += 32) {
    v16bf b  = frag_f32(Bm, R_, o0, k0, lane);
    v16bf a0 = frag_bf16(XA, R_, m0,      k0, lane);
    v16bf a1 = frag_bf16(XA, R_, m0 + 16, k0, lane);
    v16bf a2 = frag_bf16(XA, R_, m0 + 32, k0, lane);
    v16bf a3 = frag_bf16(XA, R_, m0 + 48, k0, lane);
    lacc0 = wmma_bf16(a0, b, lacc0);
    lacc1 = wmma_bf16(a1, b, lacc1);
    lacc2 = wmma_bf16(a2, b, lacc2);
    lacc3 = wmma_bf16(a3, b, lacc3);
  }
  const int n = o0 + (lane & 15);
  v8f accs[4]  = {acc0, acc1, acc2, acc3};
  v8f laccs[4] = {lacc0, lacc1, lacc2, lacc3};
#pragma unroll
  for (int mt = 0; mt < 4; ++mt) {
    const int mrow = m0 + mt * 16 + 8 * (lane >> 4);
#pragma unroll
    for (int r = 0; r < 8; ++r)
      Y[(long)(mrow + r) * O + n] = accs[mt][r] + SCALE_ * laccs[mt][r];
  }
}

// RoPE on [N, nh*128] f32 -> bf16 (half-split rotation per reference)
__global__ __launch_bounds__(64) void k_rope(const float* __restrict__ Q,
                                             bf16* __restrict__ Out, int nh) {
  const int idx = blockIdx.x;
  const long tok = idx / nh;
  const int head = idx % nh;
  const int pos = (int)(tok % S_);
  const int i = threadIdx.x;  // 0..63
  const float inv = __expf(-__logf(10000.f) * (float)i * (1.f / 64.f));
  const float ang = (float)pos * inv;
  const float c = __cosf(ang), sn = __sinf(ang);
  const float* q = Q + tok * (long)nh * HD_ + (long)head * HD_;
  bf16* o = Out + tok * (long)nh * HD_ + (long)head * HD_;
  const float x1 = q[i], x2 = q[i + 64];
  o[i]      = (bf16)(x1 * c - x2 * sn);
  o[i + 64] = (bf16)(x2 * c + x1 * sn);
}

__global__ __launch_bounds__(256) void k_cast_bf16(const float* __restrict__ in,
                                                   bf16* __restrict__ out,
                                                   long n) {
  long i = (long)blockIdx.x * 256 + threadIdx.x;
  if (i < n) out[i] = (bf16)in[i];
}

// Flash-style GQA attention: one workgroup per (b,h), 8 waves x 16 query rows.
__global__ __launch_bounds__(256) void k_attention(
    const bf16* __restrict__ Qb, const bf16* __restrict__ Kb,
    const bf16* __restrict__ Vb, const int* __restrict__ mask,
    bf16* __restrict__ Ctx) {
  const int bh = blockIdx.x;
  const int b = bh / H_, h = bh % H_;
  const int hk = h / (H_ / HK_);
  const int lane = threadIdx.x & 31, wave = threadIdx.x >> 5;
  const int q0 = wave * 16;
  const int half = lane >> 4, nl = lane & 15;

  const bf16* Qbase = Qb + ((long)b * S_) * (H_ * HD_) + (long)h * HD_;
  const bf16* Kbase = Kb + ((long)b * S_) * KVD_ + (long)hk * HD_;
  const bf16* Vbase = Vb + ((long)b * S_) * KVD_ + (long)hk * HD_;

  // ---- scores = Q @ K^T / sqrt(HD) ----
  v8f sc[8];
#pragma unroll
  for (int t = 0; t < 8; ++t) sc[t] = (v8f){};
#pragma unroll
  for (int t = 0; t < 8; ++t)
    for (int k0 = 0; k0 < HD_; k0 += 32) {
      v16bf a = frag_bf16(Qbase, H_ * HD_, q0, k0, lane);
      v16bf kfrag = frag_bf16(Kbase, KVD_, t * 16, k0, lane);  // row of K == col of K^T
      sc[t] = wmma_bf16(a, kfrag, sc[t]);
    }

  // ---- mask + numerically stable softmax (rows split over 16 lanes) ----
  const float isq = 0.08838834764831845f;  // 1/sqrt(128)
#pragma unroll
  for (int r = 0; r < 8; ++r) {
    const int qrow = q0 + r + 8 * half;
    float mx = -3.4e38f;
#pragma unroll
    for (int t = 0; t < 8; ++t) {
      const int key = t * 16 + nl;
      float v = sc[t][r] * isq;
      const bool keep = (key <= qrow) && (mask[b * S_ + key] != 0);
      v = keep ? v : -1e9f;
      sc[t][r] = v;
      mx = fmaxf(mx, v);
    }
    for (int d = 1; d < 16; d <<= 1) mx = fmaxf(mx, __shfl_xor(mx, d, 16));
    float sum = 0.f;
#pragma unroll
    for (int t = 0; t < 8; ++t) {
      float e = __expf(sc[t][r] - mx);
      sc[t][r] = e;
      sum += e;
    }
    for (int d = 1; d < 16; d <<= 1) sum += __shfl_xor(sum, d, 16);
    const float rinv = 1.f / sum;
#pragma unroll
    for (int t = 0; t < 8; ++t) sc[t][r] *= rinv;
  }

  // ---- round-trip probs through LDS to re-fragment as WMMA-A ----
  __shared__ bf16 lds_attn[8][16][128];
#pragma unroll
  for (int t = 0; t < 8; ++t)
#pragma unroll
    for (int r = 0; r < 8; ++r)
      lds_attn[wave][r + 8 * half][t * 16 + nl] = (bf16)sc[t][r];
  __syncthreads();

  // ---- ctx = P @ V ----
  for (int td = 0; td < 8; ++td) {
    v8f acc = {};
#pragma unroll
    for (int kt = 0; kt < 4; ++kt) {
      v16bf a = frag_bf16(&lds_attn[wave][0][0], 128, 0, kt * 32, lane);
      // B fragment: column (td*16+nl) of V over 32 keys (strided gather)
      v16bf bf;
      const int dcol = td * 16 + nl;
      const int kbase = kt * 32 + 8 * half;
#pragma unroll
      for (int e = 0; e < 8; ++e) {
        bf[e]     = Vbase[(long)(kbase + e) * KVD_ + dcol];
        bf[e + 8] = Vbase[(long)(kbase + 16 + e) * KVD_ + dcol];
      }
      acc = wmma_bf16(a, bf, acc);
    }
#pragma unroll
    for (int r = 0; r < 8; ++r) {
      const int qrow = q0 + r + 8 * half;
      Ctx[(((long)b * S_ + qrow) * H_ + h) * HD_ + td * 16 + nl] = (bf16)acc[r];
    }
  }
}

__global__ __launch_bounds__(256) void k_add(float* __restrict__ h,
                                             const float* __restrict__ y,
                                             long n) {
  long i = (long)blockIdx.x * 256 + threadIdx.x;
  if (i < n) h[i] += y[i];
}

__global__ __launch_bounds__(256) void k_silu_mul(const float* __restrict__ g,
                                                  const float* __restrict__ u,
                                                  bf16* __restrict__ out,
                                                  long n) {
  long i = (long)blockIdx.x * 256 + threadIdx.x;
  if (i < n) {
    float gv = g[i];
    out[i] = (bf16)((gv / (1.f + __expf(-gv))) * u[i]);
  }
}

__global__ __launch_bounds__(256) void k_final(const float* __restrict__ h,
                                               const int* __restrict__ mask,
                                               const float* __restrict__ w,
                                               float* __restrict__ out) {
  const int b = blockIdx.x;
  __shared__ float red[256];
  // last = sum(mask[b]) - 1
  int cnt = 0;
  for (int i = threadIdx.x; i < S_; i += 256) cnt += (mask[b * S_ + i] != 0);
  red[threadIdx.x] = (float)cnt;
  __syncthreads();
  for (int off = 128; off > 0; off >>= 1) {
    if (threadIdx.x < off) red[threadIdx.x] += red[threadIdx.x + off];
    __syncthreads();
  }
  const int last = (int)red[0] - 1;
  const float* x = h + ((long)b * S_ + last) * D_;
  __syncthreads();
  float s = 0.f;
  for (int i = threadIdx.x; i < D_; i += 256) { float v = x[i]; s += v * v; }
  red[threadIdx.x] = s;
  __syncthreads();
  for (int off = 128; off > 0; off >>= 1) {
    if (threadIdx.x < off) red[threadIdx.x] += red[threadIdx.x + off];
    __syncthreads();
  }
  const float inv = rsqrtf(red[0] / (float)D_ + EPS_);
  for (int i = threadIdx.x; i < D_; i += 256)
    out[(long)b * D_ + i] = x[i] * inv * w[i];
}

// ---------------------------------------------------------------------------
// Host orchestration
// ---------------------------------------------------------------------------

extern "C" void kernel_launch(void* const* d_in, const int* in_sizes, int n_in,
                              void* d_out, int out_size, void* d_ws,
                              size_t ws_size, hipStream_t stream) {
  (void)in_sizes; (void)n_in; (void)out_size; (void)ws_size;

  const int*   ids      = (const int*)d_in[0];
  const int*   amask    = (const int*)d_in[1];
  const float* emb      = (const float*)d_in[2];
  const float* attn_nw  = (const float*)d_in[3];
  const float* mlp_nw   = (const float*)d_in[4];
  const float* final_nw = (const float*)d_in[5];

  // per-projection parameter blocks: codes, absmax, A, B at indices 6..33
  struct Proj { const int* codes; const float* absmax; const float* A; const float* Bm; int O; int K; };
  auto proj = [&](int base, int O, int K) -> Proj {
    return Proj{(const int*)d_in[base], (const float*)d_in[base + 1],
                (const float*)d_in[base + 2], (const float*)d_in[base + 3], O, K};
  };
  const Proj Pq = proj(6, D_, D_),   Pk = proj(10, KVD_, D_),
             Pv = proj(14, KVD_, D_), Po = proj(18, D_, D_),
             Pg = proj(22, F_, D_),  Pu = proj(26, F_, D_),
             Pd = proj(30, D_, F_);

  // workspace carve-up
  char* ws = (char*)d_ws;
  float* h    = (float*)ws; ws += (size_t)N_ * D_ * 4;
  bf16*  xb   = (bf16*)ws;  ws += (size_t)N_ * F_ * 2;   // activations (bf16)
  bf16*  xa   = (bf16*)ws;  ws += (size_t)N_ * R_ * 2;   // LoRA XA
  float* bufA = (float*)ws; ws += (size_t)N_ * F_ * 4;   // gemm out A
  float* bufB = (float*)ws; ws += (size_t)N_ * F_ * 4;   // gemm out B
  bf16*  qb   = (bf16*)ws;  ws += (size_t)N_ * D_ * 2;   // q after rope
  bf16*  kb   = (bf16*)ws;  ws += (size_t)N_ * KVD_ * 2; // k after rope
  bf16*  vb   = (bf16*)ws;  ws += (size_t)N_ * KVD_ * 2; // v bf16
  float* vf   = bufB + (size_t)N_ * KVD_;                // v f32 (inside bufB)

  const dim3 blk128(128), blk256(256), blk64(64);
  auto run_qlinear = [&](const Proj& P, int l, const bf16* X, float* Y) {
    const long Asz = (long)R_ * P.K, Bsz = (long)P.O * R_;
    const long Csz = (long)P.O * P.K, Msz = (long)P.O * (P.K / 64);
    if (P.K == D_) {
      k_lora_xa<D_><<<dim3(N_ / 64), blk128, 0, stream>>>(X, P.A + (long)l * Asz, xa);
      k_qlinear<D_><<<dim3(P.O / 64, N_ / 64), blk128, 0, stream>>>(
          X, P.codes + (long)l * Csz, P.absmax + (long)l * Msz, xa,
          P.Bm + (long)l * Bsz, Y, P.O);
    } else {
      k_lora_xa<F_><<<dim3(N_ / 64), blk128, 0, stream>>>(X, P.A + (long)l * Asz, xa);
      k_qlinear<F_><<<dim3(P.O / 64, N_ / 64), blk128, 0, stream>>>(
          X, P.codes + (long)l * Csz, P.absmax + (long)l * Msz, xa,
          P.Bm + (long)l * Bsz, Y, P.O);
    }
  };

  k_embed<<<N_, blk256, 0, stream>>>(ids, emb, h);

  for (int l = 0; l < L_; ++l) {
    // ---- attention block ----
    k_rmsnorm<<<N_, blk256, 0, stream>>>(h, attn_nw + (long)l * D_, xb);
    run_qlinear(Pq, l, xb, bufA);                    // q f32 [N,D]
    run_qlinear(Pk, l, xb, bufB);                    // k f32 [N,512]
    run_qlinear(Pv, l, xb, vf);                      // v f32 [N,512]
    k_rope<<<N_ * H_,  blk64, 0, stream>>>(bufA, qb, H_);
    k_rope<<<N_ * HK_, blk64, 0, stream>>>(bufB, kb, HK_);
    k_cast_bf16<<<(N_ * KVD_) / 256, blk256, 0, stream>>>(vf, vb, (long)N_ * KVD_);
    k_attention<<<B_ * H_, blk256, 0, stream>>>(qb, kb, vb, amask, xb);  // ctx -> xb bf16 [N,D]
    run_qlinear(Po, l, xb, bufA);
    k_add<<<((long)N_ * D_) / 256, blk256, 0, stream>>>(h, bufA, (long)N_ * D_);

    // ---- MLP block ----
    k_rmsnorm<<<N_, blk256, 0, stream>>>(h, mlp_nw + (long)l * D_, xb);
    run_qlinear(Pg, l, xb, bufA);                    // gate f32 [N,F]
    run_qlinear(Pu, l, xb, bufB);                    // up   f32 [N,F]
    k_silu_mul<<<((long)N_ * F_) / 256, blk256, 0, stream>>>(bufA, bufB, xb, (long)N_ * F_);
    run_qlinear(Pd, l, xb, bufA);                    // down f32 [N,D]
    k_add<<<((long)N_ * D_) / 256, blk256, 0, stream>>>(h, bufA, (long)N_ * D_);
  }

  k_final<<<B_, blk256, 0, stream>>>(h, amask, final_nw, (float*)d_out);
}
